// ResidualMP_72610717106485
// MI455X (gfx1250) — matrix operand order repaired and checked by reading, not compiled, not run on hardware
//
#include <hip/hip_runtime.h>
#include <hip/hip_bf16.h>
#include <math.h>

#define N_NODES 100000
#define N_EDGES 1600000
#define DIM     128
#define NLAYER  3
#define NOUT    47
#define LDH     136   // padded f16 LDS row stride (halves): 272B = 17*16B -> conflict-free, 16B aligned

typedef __attribute__((ext_vector_type(16))) _Float16 v16h;
typedef __attribute__((ext_vector_type(8)))  _Float16 v8h;
typedef __attribute__((ext_vector_type(8)))  float    v8f;

// ---------------------------------------------------------------------------
// WMMA fragment loaders (CDNA5 wave32 layouts, CDNA5 ISA 7.12.2)
// ---------------------------------------------------------------------------

// A: 16x32 f16 tile from an f16 LDS strip (row-major, leading dim LDH halves).
// lane L: row = L&15 ; K base = k0 + (L>=16 ? 8 : 0); halves 0..7 = K..K+7,
// halves 8..15 = K+16..K+23  ->  two 16-byte ds loads.
__device__ __forceinline__ v16h load_A_ldsh(const _Float16* Ah, int k0, int lane) {
    int m  = lane & 15;
    int kk = k0 + ((lane >> 4) << 3);
    const _Float16* p = Ah + m * LDH + kk;
    v8h lo = *(const v8h*)(p);
    v8h hi = *(const v8h*)(p + 16);
    v16h a;
#pragma unroll
    for (int i = 0; i < 8; ++i) { a[i] = lo[i]; a[8 + i] = hi[i]; }
    return a;
}

// B: 32x16 f16 tile for D = A * W^T : B[k][n] = W[n][k].
// lane L: col n (passed in), K base = k0 + (L>=16 ? 8 : 0); halves 0..7 = K..K+7,
// halves 8..15 = K+16..K+23 of row n (matches A).
__device__ __forceinline__ v16h load_B_w(const _Float16* Wh, int n, int k0, int lane) {
    int kk = k0 + ((lane >> 4) << 3);
    const _Float16* p = Wh + (size_t)n * DIM + kk;
    v8h lo = *(const v8h*)(p);
    v8h hi = *(const v8h*)(p + 16);
    v16h b;
#pragma unroll
    for (int i = 0; i < 8; ++i) { b[i] = lo[i]; b[8 + i] = hi[i]; }
    return b;
}

// Full K=128 strip: preload 4 A + 4 B fragments, then 4 chained WMMAs.
__device__ __forceinline__ v8f wmma_strip(const _Float16* Ah, const _Float16* Wh,
                                          int n, int lane) {
    v16h b0 = load_B_w(Wh, n,  0, lane);
    v16h b1 = load_B_w(Wh, n, 32, lane);
    v16h b2 = load_B_w(Wh, n, 64, lane);
    v16h b3 = load_B_w(Wh, n, 96, lane);
    v16h a0 = load_A_ldsh(Ah,  0, lane);
    v16h a1 = load_A_ldsh(Ah, 32, lane);
    v16h a2 = load_A_ldsh(Ah, 64, lane);
    v16h a3 = load_A_ldsh(Ah, 96, lane);
    v8f c = {};
    c = __builtin_amdgcn_wmma_f32_16x16x32_f16(false, a0, false, b0, (short)0, c, false, false);
    c = __builtin_amdgcn_wmma_f32_16x16x32_f16(false, a1, false, b1, (short)0, c, false, false);
    c = __builtin_amdgcn_wmma_f32_16x16x32_f16(false, a2, false, b2, (short)0, c, false, false);
    c = __builtin_amdgcn_wmma_f32_16x16x32_f16(false, a3, false, b3, (short)0, c, false, false);
    return c;
}

// Stage a 16xDIM fp32 strip into the padded f16 LDS tile (converts once,
// shared by all 8 waves). nthreads = blockDim.x.
__device__ __forceinline__ void stage_strip_f16(_Float16* Ah, const float* In,
                                                int mtile, int nthreads) {
    for (int idx = threadIdx.x; idx < 16 * (DIM / 2); idx += nthreads) {
        int r = idx >> 6, c2 = (idx & 63) * 2;
        const float* s = In + (size_t)(mtile + r) * DIM + c2;
        Ah[r * LDH + c2]     = (_Float16)s[0];
        Ah[r * LDH + c2 + 1] = (_Float16)s[1];
    }
}

// ---------------------------------------------------------------------------
// Weight fp32 -> f16 conversion (once per launch)
// layout: [0,3NW) Wl, [3NW,6NW) Wr, [6NW,7NW) W1, [7NW, 7NW+48*128) W2 padded
// ---------------------------------------------------------------------------
__global__ void cvt_weights_kernel(const float* Wl, const float* Wr,
                                   const float* W1, const float* W2, _Float16* WH) {
    const int NW = DIM * DIM;
    int i = blockIdx.x * 256 + threadIdx.x;
    if (i < 3 * NW)            WH[i] = (_Float16)Wl[i];
    else if (i < 6 * NW)       WH[i] = (_Float16)Wr[i - 3 * NW];
    else if (i < 7 * NW)       WH[i] = (_Float16)W1[i - 6 * NW];
    else if (i < 7 * NW + 48 * DIM) {
        int j = i - 7 * NW;
        WH[i] = (j < NOUT * DIM) ? (_Float16)W2[j] : (_Float16)0.0f;
    }
}

// ---------------------------------------------------------------------------
// Edge scatter-sum: one wave per edge, float4 per lane, f32 global atomics.
// Prefetch a future edge's source row (random-access gather is latency bound).
// ---------------------------------------------------------------------------
__global__ void scatter_kernel(const float* __restrict__ X,
                               const int* __restrict__ src,
                               const int* __restrict__ dst,
                               float* __restrict__ Z, int nEdges) {
    int e = blockIdx.x * 8 + (threadIdx.x >> 5);
    if (e >= nEdges) return;
    int lane = threadIdx.x & 31;
    int ePF = e + 2048;
    if (ePF < nEdges) {
        int sp = src[ePF];
        __builtin_prefetch(X + (size_t)sp * DIM + lane * 4, 0, 1);  // global_prefetch_b8
    }
    int s = src[e], d = dst[e];
    float4 v = ((const float4*)(X + (size_t)s * DIM))[lane];
    float* zp = Z + (size_t)d * DIM + lane * 4;
    atomicAdd(zp + 0, v.x);
    atomicAdd(zp + 1, v.y);
    atomicAdd(zp + 2, v.z);
    atomicAdd(zp + 3, v.w);
}

// ---------------------------------------------------------------------------
// GEMM: Out = In @ W^T + b  (optionally ReLU + per-feature sum/sumsq atomics)
// 256 threads = 8 waves; block owns 16 rows x 128 cols; wave w -> cols 16w..
// ---------------------------------------------------------------------------
template <bool RELU_STATS>
__global__ void gemm_kernel(const float* __restrict__ In, const _Float16* __restrict__ Wh,
                            const float* __restrict__ bias,
                            float* __restrict__ Out, float* __restrict__ stats) {
    __shared__ __align__(16) _Float16 Ah[16 * LDH];
    int mtile = blockIdx.x * 16;
    stage_strip_f16(Ah, In, mtile, 256);
    __syncthreads();

    int wave = threadIdx.x >> 5, lane = threadIdx.x & 31;
    int n = wave * 16 + (lane & 15);
    v8f c = wmma_strip(Ah, Wh, n, lane);

    float b = bias[n];
    int mbase = mtile + ((lane >> 4) << 3);
    if (RELU_STATS) {
        float s = 0.f, s2 = 0.f;
#pragma unroll
        for (int j = 0; j < 8; ++j) {
            float v = c[j] + b;
            v = v > 0.f ? v : 0.f;
            Out[(size_t)(mbase + j) * DIM + n] = v;
            s += v; s2 += v * v;
        }
        atomicAdd(&stats[n], s);
        atomicAdd(&stats[DIM + n], s2);
    } else {
#pragma unroll
        for (int j = 0; j < 8; ++j)
            Out[(size_t)(mbase + j) * DIM + n] = c[j] + b;
    }
}

// ---------------------------------------------------------------------------
// BN stats finalize: scale = gamma*rstd, shift = beta - mu*scale
// ---------------------------------------------------------------------------
__global__ void stats_kernel(const float* __restrict__ stats,
                             const float* __restrict__ gamma,
                             const float* __restrict__ beta,
                             float* __restrict__ scale, float* __restrict__ shift) {
    int n = threadIdx.x;
    const float invN = 1.0f / (float)N_NODES;
    float mu  = stats[n] * invN;
    float var = stats[DIM + n] * invN - mu * mu;
    float rstd = rsqrtf(var + 1e-5f);
    float sc = gamma[n] * rstd;
    scale[n] = sc;
    shift[n] = beta[n] - mu * sc;
}

// ---------------------------------------------------------------------------
// Combine: out = X@Wl^T + bl + (H*scale + shift) + Z ; row L2-norm ; ReLU
// ---------------------------------------------------------------------------
__global__ void gemm_combine_kernel(const float* __restrict__ X, const _Float16* __restrict__ Wh,
                                    const float* __restrict__ bl,
                                    const float* __restrict__ H, const float* __restrict__ Z,
                                    const float* __restrict__ scale, const float* __restrict__ shift,
                                    float* __restrict__ Xout) {
    __shared__ __align__(16) _Float16 Ah[16 * LDH];
    __shared__ float Otile[16 * 132];
    __shared__ float rpart[16][17];
    __shared__ float rnorm[16];
    int mtile = blockIdx.x * 16;
    stage_strip_f16(Ah, X, mtile, 256);
    __syncthreads();

    int wave = threadIdx.x >> 5, lane = threadIdx.x & 31;
    int n = wave * 16 + (lane & 15);
    v8f c = wmma_strip(Ah, Wh, n, lane);

    float b = bl[n], sc = scale[n], sh = shift[n];
    int mloc = (lane >> 4) << 3;
#pragma unroll
    for (int j = 0; j < 8; ++j) {
        size_t g = (size_t)(mtile + mloc + j) * DIM + n;
        float v = c[j] + b + H[g] * sc + sh + Z[g];
        Otile[(mloc + j) * 132 + n] = v;
    }
    __syncthreads();

    {   // per-row sum of squares: 16 threads per row, 8 cols each
        int row = threadIdx.x >> 4, seg = threadIdx.x & 15;
        const float* p = Otile + row * 132 + seg * 8;
        float s = 0.f;
#pragma unroll
        for (int i = 0; i < 8; ++i) s += p[i] * p[i];
        rpart[row][seg] = s;
    }
    __syncthreads();
    if (threadIdx.x < 16) {
        float s = 0.f;
#pragma unroll
        for (int j = 0; j < 16; ++j) s += rpart[threadIdx.x][j];
        rnorm[threadIdx.x] = 1.0f / fmaxf(sqrtf(s), 1e-12f);
    }
    __syncthreads();
    for (int idx = threadIdx.x; idx < 16 * DIM; idx += 256) {
        int r = idx >> 7, cc = idx & 127;
        float v = Otile[r * 132 + cc] * rnorm[r];
        Xout[(size_t)(mtile + r) * DIM + cc] = fmaxf(v, 0.f);
    }
}

// ---------------------------------------------------------------------------
// Final projection to 47 classes + log_softmax. 128 threads; waves 0..2 do
// three 16-col WMMA tiles (padded to 48); 16 threads finish softmax rows.
// ---------------------------------------------------------------------------
__global__ void post2_kernel(const float* __restrict__ Mid, const _Float16* __restrict__ W2h,
                             const float* __restrict__ b2, float* __restrict__ Out) {
    __shared__ __align__(16) _Float16 Ah[16 * LDH];
    __shared__ float Otile[16 * 48];
    int mtile = blockIdx.x * 16;
    stage_strip_f16(Ah, Mid, mtile, 128);
    __syncthreads();

    int wave = threadIdx.x >> 5, lane = threadIdx.x & 31;
    if (wave < 3) {
        int n = wave * 16 + (lane & 15);
        v8f c = wmma_strip(Ah, W2h, n, lane);
        float b = (n < NOUT) ? b2[n] : 0.f;
        int mloc = (lane >> 4) << 3;
#pragma unroll
        for (int j = 0; j < 8; ++j)
            Otile[(mloc + j) * 48 + n] = c[j] + b;
    }
    __syncthreads();
    if (threadIdx.x < 16) {
        const float* r = Otile + threadIdx.x * 48;
        float mx = -1e30f;
        for (int k = 0; k < NOUT; ++k) mx = fmaxf(mx, r[k]);
        float se = 0.f;
        for (int k = 0; k < NOUT; ++k) se += __expf(r[k] - mx);
        float lse = mx + __logf(se);
        float* o = Out + (size_t)(mtile + threadIdx.x) * NOUT;
        for (int k = 0; k < NOUT; ++k) o[k] = r[k] - lse;
    }
}

// ---------------------------------------------------------------------------
// Host launcher
// ---------------------------------------------------------------------------
extern "C" void kernel_launch(void* const* d_in, const int* in_sizes, int n_in,
                              void* d_out, int out_size, void* d_ws, size_t ws_size,
                              hipStream_t stream) {
    const float* x     = (const float*)d_in[0];
    const int*   ei    = (const int*)d_in[1];
    const float* Wl    = (const float*)d_in[2];
    const float* bl    = (const float*)d_in[3];
    const float* Wr    = (const float*)d_in[4];
    const float* br    = (const float*)d_in[5];
    const float* gamma = (const float*)d_in[6];
    const float* beta  = (const float*)d_in[7];
    const float* W1    = (const float*)d_in[8];
    const float* b1    = (const float*)d_in[9];
    const float* W2    = (const float*)d_in[10];
    const float* b2    = (const float*)d_in[11];
    float* out = (float*)d_out;

    const size_t NF = (size_t)N_NODES * DIM;
    float* Z      = (float*)d_ws;
    float* H      = Z + NF;
    float* XA     = H + NF;
    float* XB     = XA + NF;
    float* stats  = XB + NF;     // 256 floats (sum, sumsq)
    float* scale  = stats + 256; // 128
    float* shiftv = scale + DIM; // 128
    _Float16* WH  = (_Float16*)(shiftv + DIM);
    _Float16* WlH = WH;
    _Float16* WrH = WlH + 3 * DIM * DIM;
    _Float16* W1H = WrH + 3 * DIM * DIM;
    _Float16* W2H = W1H + DIM * DIM;     // 48x128 padded

    const int NWT = 7 * DIM * DIM + 48 * DIM;   // 120832 halves
    cvt_weights_kernel<<<(NWT + 255) / 256, 256, 0, stream>>>(Wl, Wr, W1, W2, WH);

    const int* srcI = ei;
    const int* dstI = ei + N_EDGES;
    const int nStrips = N_NODES / 16;           // 6250 (exact)

    const float* xin = x;
    float* xout = XA;
    for (int l = 0; l < NLAYER; ++l) {
        hipMemsetAsync(Z, 0, NF * sizeof(float), stream);
        hipMemsetAsync(stats, 0, 256 * sizeof(float), stream);
        scatter_kernel<<<N_EDGES / 8, 256, 0, stream>>>(xin, srcI, dstI, Z, N_EDGES);
        gemm_kernel<true><<<nStrips, 256, 0, stream>>>(Z, WrH + (size_t)l * DIM * DIM,
                                                       br + l * DIM, H, stats);
        stats_kernel<<<1, DIM, 0, stream>>>(stats, gamma + l * DIM, beta + l * DIM,
                                            scale, shiftv);
        gemm_combine_kernel<<<nStrips, 256, 0, stream>>>(xin, WlH + (size_t)l * DIM * DIM,
                                                         bl + l * DIM, H, Z, scale, shiftv,
                                                         xout);
        xin = xout;
        xout = (xout == XA) ? XB : XA;
    }

    float* Mid = Z;  // reuse
    gemm_kernel<false><<<nStrips, 256, 0, stream>>>(xin, W1H, b1, Mid, nullptr);
    post2_kernel<<<nStrips, 128, 0, stream>>>(Mid, W2H, b2, out);
}
